// Lstm_9878424781322
// MI455X (gfx1250) — matrix operand order repaired and checked by reading, not compiled, not run on hardware
//
#include <hip/hip_runtime.h>

// LSTM cell on MI455X (gfx1250, wave32, WMMA):
//   Path A (if ws fits): one-shot fp32->bf16 conversion of concat(i,h) and W1..W4
//     into workspace (bf16 working set = 134MB, fits 192MB L2), then a fused
//     4-gate bf16 WMMA GEMM + gate/cell epilogue with double-buffered LDS.
//   Path B (fallback): single fused kernel converting on the fly.

typedef __attribute__((ext_vector_type(16))) __bf16 v16bf;
typedef __attribute__((ext_vector_type(8)))  float  v8f;
typedef __attribute__((ext_vector_type(4)))  float  f4;

#define BATCH   8192
#define IN_DIM  2048
#define OUT_DIM 2048
#define K_DIM   4096

#define BM 128          // block tile M (batch)
#define BN 64           // block tile N (output features)
#define BK 64           // K chunk per LDS stage
#define LDT (BK + 8)    // LDS row stride in bf16 elems (padded, keeps 16B align)
#define CHUNK (BM * LDT + 4 * BN * LDT)   // ushorts per LDS buffer (27648)

union Frag {
    v16bf v;
    uint4 q[2];
};

__device__ __forceinline__ unsigned int f2bf(float f) {
    unsigned int u = __float_as_uint(f);
    return (u + 0x7FFFu + ((u >> 16) & 1u)) >> 16;   // round-to-nearest-even
}

__device__ __forceinline__ float sigmoidf(float x) {
    return 1.0f / (1.0f + __expf(-x));
}

// ---------------------------------------------------------------------------
// Conversion kernels (run once per launch): fp32 -> bf16 into workspace.
// ---------------------------------------------------------------------------
__device__ __forceinline__ uint4 cvt8(const float* __restrict__ src) {
    f4 a = __builtin_nontemporal_load(reinterpret_cast<const f4*>(src));
    f4 b = __builtin_nontemporal_load(reinterpret_cast<const f4*>(src + 4));
    uint4 o;
    o.x = f2bf(a.x) | (f2bf(a.y) << 16);
    o.y = f2bf(a.z) | (f2bf(a.w) << 16);
    o.z = f2bf(b.x) | (f2bf(b.y) << 16);
    o.w = f2bf(b.z) | (f2bf(b.w) << 16);
    return o;
}

__global__ void convert_ih(unsigned short* __restrict__ dst,   // [8192][4096] bf16
                           const float* __restrict__ I,
                           const float* __restrict__ H)
{
    int idx = blockIdx.x * 256 + threadIdx.x;     // 4,194,304 threads, 8 elems each
    size_t base = (size_t)idx * 8;
    int b   = (int)(base >> 12);
    int col = (int)(base & 4095);
    const float* src = (col < IN_DIM) ? (I + (size_t)b * IN_DIM + col)
                                      : (H + (size_t)b * OUT_DIM + (col - IN_DIM));
    *reinterpret_cast<uint4*>(dst + base) = cvt8(src);
}

__global__ void convert_w(unsigned short* __restrict__ dst,    // [4][2048][4096] bf16
                          const float* __restrict__ W1, const float* __restrict__ W2,
                          const float* __restrict__ W3, const float* __restrict__ W4)
{
    int idx = blockIdx.x * 256 + threadIdx.x;
    size_t base = (size_t)idx * 8;
    int    g   = (int)(base >> 23);               // 2048*4096 = 1<<23 elems per gate
    size_t rem = base & ((1u << 23) - 1);
    const float* src = (g == 0) ? W1 : (g == 1) ? W2 : (g == 2) ? W3 : W4;
    *reinterpret_cast<uint4*>(dst + base) = cvt8(src + rem);
}

// ---------------------------------------------------------------------------
// Main GEMM (bf16 pre-converted), double-buffered LDS pipeline.
// ---------------------------------------------------------------------------
__launch_bounds__(256, 1)
__global__ void lstm_gemm_bf16(const unsigned short* __restrict__ ihB,  // [8192][4096]
                               const unsigned short* __restrict__ Wb,   // [4][2048][4096]
                               const float* __restrict__ C,
                               const float* __restrict__ B1, const float* __restrict__ B2,
                               const float* __restrict__ B3, const float* __restrict__ B4,
                               float* __restrict__ out)
{
    extern __shared__ unsigned short smem[];      // 2 * CHUNK ushorts = 110592 B

    const int tid    = threadIdx.x;
    const int lane   = tid & 31;
    const int wave   = tid >> 5;
    const int waveM  = wave >> 1;
    const int waveN  = wave & 1;
    const int blockM = blockIdx.y * BM;
    const int blockN = blockIdx.x * BN;
    const int l15    = lane & 15;
    const int lhi    = lane >> 4;

    v8f acc[4][2][2];
#pragma unroll
    for (int g = 0; g < 4; ++g)
#pragma unroll
        for (int sm = 0; sm < 2; ++sm)
#pragma unroll
            for (int sn = 0; sn < 2; ++sn)
#pragma unroll
                for (int r = 0; r < 8; ++r)
                    acc[g][sm][sn][r] = 0.0f;

    uint4 pa[4];   // A prefetch: 128x64 bf16 / 256 thr = 4 x uint4 (8 bf16 each)
    uint4 pw[8];   // W prefetch: 4x64x64 bf16 / 256 thr = 8 x uint4

    // ---- prologue: fetch + stage chunk 0 into buffer 0 ----
#pragma unroll
    for (int r = 0; r < 4; ++r) {
        int idx = tid + r * 256, row = idx >> 3, c8 = idx & 7;
        pa[r] = *reinterpret_cast<const uint4*>(
            ihB + (size_t)(blockM + row) * K_DIM + c8 * 8);
    }
#pragma unroll
    for (int r = 0; r < 8; ++r) {
        int idx = tid + r * 256, g = idx >> 9, rem = idx & 511;
        int row = rem >> 3, c8 = rem & 7;
        pw[r] = *reinterpret_cast<const uint4*>(
            Wb + ((size_t)g << 23) + (size_t)(blockN + row) * K_DIM + c8 * 8);
    }
    {
        unsigned short* buf = smem;
#pragma unroll
        for (int r = 0; r < 4; ++r) {
            int idx = tid + r * 256, row = idx >> 3, c8 = idx & 7;
            *reinterpret_cast<uint4*>(buf + row * LDT + c8 * 8) = pa[r];
        }
#pragma unroll
        for (int r = 0; r < 8; ++r) {
            int idx = tid + r * 256, g = idx >> 9, rem = idx & 511;
            int row = rem >> 3, c8 = rem & 7;
            *reinterpret_cast<uint4*>(buf + BM * LDT + g * (BN * LDT)
                                      + row * LDT + c8 * 8) = pw[r];
        }
    }

    const int NCHUNK = K_DIM / BK;   // 64
    for (int kc = 0; kc < NCHUNK; ++kc) {
        __syncthreads();             // chunk kc staged; prior readers of (kc+1)&1 done

        // issue global loads for chunk kc+1 (latency hides under WMMA chain)
        if (kc + 1 < NCHUNK) {
            const int kk = (kc + 1) * BK;
#pragma unroll
            for (int r = 0; r < 4; ++r) {
                int idx = tid + r * 256, row = idx >> 3, c8 = idx & 7;
                pa[r] = *reinterpret_cast<const uint4*>(
                    ihB + (size_t)(blockM + row) * K_DIM + kk + c8 * 8);
            }
#pragma unroll
            for (int r = 0; r < 8; ++r) {
                int idx = tid + r * 256, g = idx >> 9, rem = idx & 511;
                int row = rem >> 3, c8 = rem & 7;
                pw[r] = *reinterpret_cast<const uint4*>(
                    Wb + ((size_t)g << 23) + (size_t)(blockN + row) * K_DIM + kk + c8 * 8);
            }
        }

        // ---- compute on buffer kc&1 ----
        const unsigned short* a0 = smem + (kc & 1) * CHUNK;
        const unsigned short* w0 = a0 + BM * LDT;
#pragma unroll
        for (int ks = 0; ks < BK; ks += 32) {
            Frag af[2];
#pragma unroll
            for (int sm = 0; sm < 2; ++sm) {
                int row = waveM * 32 + sm * 16 + l15;
                int c0  = ks + (lhi << 3);
                const unsigned short* p = a0 + row * LDT + c0;
                af[sm].q[0] = *reinterpret_cast<const uint4*>(p);
                af[sm].q[1] = *reinterpret_cast<const uint4*>(p + 16);
            }
            Frag bf[8];
#pragma unroll
            for (int g = 0; g < 4; ++g)
#pragma unroll
                for (int sn = 0; sn < 2; ++sn) {
                    int col = waveN * 32 + sn * 16 + l15;
                    int c0  = ks + (lhi << 3);
                    const unsigned short* p = w0 + g * (BN * LDT) + col * LDT + c0;
                    bf[g * 2 + sn].q[0] = *reinterpret_cast<const uint4*>(p);
                    bf[g * 2 + sn].q[1] = *reinterpret_cast<const uint4*>(p + 16);
                }
#pragma unroll
            for (int g = 0; g < 4; ++g)
#pragma unroll
                for (int sn = 0; sn < 2; ++sn)
#pragma unroll
                    for (int sm = 0; sm < 2; ++sm)
                        acc[g][sm][sn] = __builtin_amdgcn_wmma_f32_16x16x32_bf16(
                            false, af[sm].v, false, bf[g * 2 + sn].v,
                            (short)0, acc[g][sm][sn], false, false);
        }

        // ---- stage chunk kc+1 into the other buffer ----
        if (kc + 1 < NCHUNK) {
            unsigned short* buf = smem + ((kc + 1) & 1) * CHUNK;
#pragma unroll
            for (int r = 0; r < 4; ++r) {
                int idx = tid + r * 256, row = idx >> 3, c8 = idx & 7;
                *reinterpret_cast<uint4*>(buf + row * LDT + c8 * 8) = pa[r];
            }
#pragma unroll
            for (int r = 0; r < 8; ++r) {
                int idx = tid + r * 256, g = idx >> 9, rem = idx & 511;
                int row = rem >> 3, c8 = rem & 7;
                *reinterpret_cast<uint4*>(buf + BM * LDT + g * (BN * LDT)
                                          + row * LDT + c8 * 8) = pw[r];
            }
        }
    }

    // ---- epilogue: gates + cell update (non-temporal c/out streams) ----
    const size_t HC = (size_t)BATCH * OUT_DIM;
#pragma unroll
    for (int sm = 0; sm < 2; ++sm) {
#pragma unroll
        for (int sn = 0; sn < 2; ++sn) {
            int n = blockN + waveN * 32 + sn * 16 + l15;
            float b1v = B1[n], b2v = B2[n], b3v = B3[n], b4v = B4[n];
#pragma unroll
            for (int r = 0; r < 8; ++r) {
                int m = blockM + waveM * 32 + sm * 16 + (lhi << 3) + r;
                size_t o = (size_t)m * OUT_DIM + n;
                float g1 = acc[0][sm][sn][r] + b1v;
                float g2 = acc[1][sm][sn][r] + b2v;
                float g3 = acc[2][sm][sn][r] + b3v;
                float g4 = acc[3][sm][sn][r] + b4v;
                float cv = __builtin_nontemporal_load(&C[o]);
                float nc = cv * sigmoidf(g1) + sigmoidf(g2) * tanhf(g3);
                float nh = tanhf(cv) * sigmoidf(g4);
                __builtin_nontemporal_store(nh, &out[o]);
                __builtin_nontemporal_store(nc, &out[HC + o]);
            }
        }
    }
}

// ---------------------------------------------------------------------------
// Fallback: fully fused kernel with on-the-fly conversion (round-1 version).
// ---------------------------------------------------------------------------
__launch_bounds__(256, 1)
__global__ void lstm_fused(const float* __restrict__ I,
                           const float* __restrict__ H,
                           const float* __restrict__ C,
                           const float* __restrict__ W1, const float* __restrict__ B1,
                           const float* __restrict__ W2, const float* __restrict__ B2,
                           const float* __restrict__ W3, const float* __restrict__ B3,
                           const float* __restrict__ W4, const float* __restrict__ B4,
                           float* __restrict__ out)
{
    __shared__ unsigned short sA[BM * LDT];
    __shared__ unsigned short sW[4][BN * LDT];

    const int tid    = threadIdx.x;
    const int lane   = tid & 31;
    const int wave   = tid >> 5;
    const int waveM  = wave >> 1;
    const int waveN  = wave & 1;
    const int blockM = blockIdx.y * BM;
    const int blockN = blockIdx.x * BN;
    const int l15    = lane & 15;
    const int lhi    = lane >> 4;

    const float* Wp[4] = {W1, W2, W3, W4};

    v8f acc[4][2][2];
#pragma unroll
    for (int g = 0; g < 4; ++g)
#pragma unroll
        for (int sm = 0; sm < 2; ++sm)
#pragma unroll
            for (int sn = 0; sn < 2; ++sn)
#pragma unroll
                for (int r = 0; r < 8; ++r)
                    acc[g][sm][sn][r] = 0.0f;

    for (int kk = 0; kk < K_DIM; kk += BK) {
        const float* aSrc    = (kk < IN_DIM) ? I : H;
        const int    colBase = kk & (IN_DIM - 1);
#pragma unroll
        for (int r = 0; r < 8; ++r) {
            int idx = tid + r * 256, row = idx >> 4, c4 = idx & 15;
            const f4 f = *reinterpret_cast<const f4*>(
                aSrc + (size_t)(blockM + row) * IN_DIM + colBase + c4 * 4);
            uint2 p;
            p.x = f2bf(f.x) | (f2bf(f.y) << 16);
            p.y = f2bf(f.z) | (f2bf(f.w) << 16);
            *reinterpret_cast<uint2*>(&sA[row * LDT + c4 * 4]) = p;
        }
#pragma unroll
        for (int g = 0; g < 4; ++g)
#pragma unroll
            for (int r = 0; r < 4; ++r) {
                int idx = tid + r * 256, row = idx >> 4, c4 = idx & 15;
                const f4 f = *reinterpret_cast<const f4*>(
                    Wp[g] + (size_t)(blockN + row) * K_DIM + kk + c4 * 4);
                uint2 p;
                p.x = f2bf(f.x) | (f2bf(f.y) << 16);
                p.y = f2bf(f.z) | (f2bf(f.w) << 16);
                *reinterpret_cast<uint2*>(&sW[g][row * LDT + c4 * 4]) = p;
            }
        __syncthreads();

#pragma unroll
        for (int ks = 0; ks < BK; ks += 32) {
            Frag af[2];
#pragma unroll
            for (int sm = 0; sm < 2; ++sm) {
                int row = waveM * 32 + sm * 16 + l15;
                int c0  = ks + (lhi << 3);
                const unsigned short* p = &sA[row * LDT + c0];
                af[sm].q[0] = *reinterpret_cast<const uint4*>(p);
                af[sm].q[1] = *reinterpret_cast<const uint4*>(p + 16);
            }
            Frag bf[8];
#pragma unroll
            for (int g = 0; g < 4; ++g)
#pragma unroll
                for (int sn = 0; sn < 2; ++sn) {
                    int col = waveN * 32 + sn * 16 + l15;
                    int c0  = ks + (lhi << 3);
                    const unsigned short* p = &sW[g][col * LDT + c0];
                    bf[g * 2 + sn].q[0] = *reinterpret_cast<const uint4*>(p);
                    bf[g * 2 + sn].q[1] = *reinterpret_cast<const uint4*>(p + 16);
                }
#pragma unroll
            for (int g = 0; g < 4; ++g)
#pragma unroll
                for (int sn = 0; sn < 2; ++sn)
#pragma unroll
                    for (int sm = 0; sm < 2; ++sm)
                        acc[g][sm][sn] = __builtin_amdgcn_wmma_f32_16x16x32_bf16(
                            false, af[sm].v, false, bf[g * 2 + sn].v,
                            (short)0, acc[g][sm][sn], false, false);
        }
        __syncthreads();
    }

    const size_t HC = (size_t)BATCH * OUT_DIM;
#pragma unroll
    for (int sm = 0; sm < 2; ++sm) {
#pragma unroll
        for (int sn = 0; sn < 2; ++sn) {
            int n = blockN + waveN * 32 + sn * 16 + l15;
            float b1v = B1[n], b2v = B2[n], b3v = B3[n], b4v = B4[n];
#pragma unroll
            for (int r = 0; r < 8; ++r) {
                int m = blockM + waveM * 32 + sm * 16 + (lhi << 3) + r;
                size_t o = (size_t)m * OUT_DIM + n;
                float g1 = acc[0][sm][sn][r] + b1v;
                float g2 = acc[1][sm][sn][r] + b2v;
                float g3 = acc[2][sm][sn][r] + b3v;
                float g4 = acc[3][sm][sn][r] + b4v;
                float cv = __builtin_nontemporal_load(&C[o]);
                float nc = cv * sigmoidf(g1) + sigmoidf(g2) * tanhf(g3);
                float nh = tanhf(cv) * sigmoidf(g4);
                __builtin_nontemporal_store(nh, &out[o]);
                __builtin_nontemporal_store(nc, &out[HC + o]);
            }
        }
    }
}

extern "C" void kernel_launch(void* const* d_in, const int* in_sizes, int n_in,
                              void* d_out, int out_size, void* d_ws, size_t ws_size,
                              hipStream_t stream) {
    (void)in_sizes; (void)n_in; (void)out_size;
    const float* I  = (const float*)d_in[0];
    const float* H  = (const float*)d_in[1];
    const float* C  = (const float*)d_in[2];
    const float* W1 = (const float*)d_in[3];
    const float* B1 = (const float*)d_in[4];
    const float* W2 = (const float*)d_in[5];
    const float* B2 = (const float*)d_in[6];
    const float* W3 = (const float*)d_in[7];
    const float* B3 = (const float*)d_in[8];
    const float* W4 = (const float*)d_in[9];
    const float* B4 = (const float*)d_in[10];
    float* out = (float*)d_out;

    const size_t ihElems = (size_t)BATCH * K_DIM;          // 33.5M
    const size_t wElems  = (size_t)4 * OUT_DIM * K_DIM;    // 33.5M
    const size_t need    = (ihElems + wElems) * sizeof(unsigned short);  // 134MB

    dim3 grid(OUT_DIM / BN, BATCH / BM);   // 32 x 64
    dim3 block(256);

    if (ws_size >= need) {
        unsigned short* ihB = (unsigned short*)d_ws;
        unsigned short* Wb  = ihB + ihElems;
        convert_ih<<<(int)(ihElems / 8 / 256), block, 0, stream>>>(ihB, I, H);
        convert_w <<<(int)(wElems  / 8 / 256), block, 0, stream>>>(Wb, W1, W2, W3, W4);
        lstm_gemm_bf16<<<grid, block, 2 * CHUNK * sizeof(unsigned short), stream>>>(
            ihB, Wb, C, B1, B2, B3, B4, out);
    } else {
        lstm_fused<<<grid, block, 0, stream>>>(I, H, C, W1, B1, W2, B2, W3, B3, W4, B4, out);
    }
}
